// AttentionalPropagation_90847148245217
// MI455X (gfx1250) — matrix operand order repaired and checked by reading, not compiled, not run on hardware
//
#include <hip/hip_runtime.h>
#include <cstdint>
#include <cstddef>

// Problem constants (match reference)
#define BB 8
#define DD 256
#define NN 2048
#define MM 2048
#define HH 4
#define DHH 64
#define D2 512

typedef __attribute__((ext_vector_type(16))) __bf16 v16bf;
typedef __attribute__((ext_vector_type(8)))  float  v8f;

union Frag16 {
    uint4          q[2];
    v16bf          v;
    unsigned short s[16];
};
static_assert(sizeof(Frag16) == 32, "frag size");

__device__ __forceinline__ unsigned short f2bf(float f) {
    union { float f; unsigned u; } c; c.f = f;
    unsigned u = c.u;
    unsigned r = u + 0x7FFFu + ((u >> 16) & 1u);
    if ((u & 0x7F800000u) == 0x7F800000u) r = u;  // inf/nan: keep payload
    return (unsigned short)(r >> 16);
}

__device__ __forceinline__ v8f wmma_bf16(const Frag16& a, const Frag16& b, v8f c) {
    return __builtin_amdgcn_wmma_f32_16x16x32_bf16(false, a.v, false, b.v,
                                                   (short)0, c, false, false);
}

// CDNA5 async global->LDS copy (16B per lane), tracked by ASYNCcnt.
// LDS operand is the 0-based workgroup LDS byte offset (low 32 bits of the
// generic pointer); global operand is a 64-bit VGPR pair.
__device__ __forceinline__ void async_load_b128(const void* gaddr, void* lds) {
    unsigned ldsoff = (unsigned)(size_t)lds;
    unsigned long long ga = (unsigned long long)(size_t)gaddr;
    asm volatile("global_load_async_to_lds_b128 %0, %1, off"
                 :: "v"(ldsoff), "v"(ga) : "memory");
}
__device__ __forceinline__ void wait_async() {
    asm volatile("s_wait_asynccnt 0" ::: "memory");
}

// ---------------------------------------------------------------------------
// fp32 -> bf16 elementwise convert
// ---------------------------------------------------------------------------
__global__ __launch_bounds__(256) void k_f32_to_bf16(
    const float* __restrict__ in, unsigned short* __restrict__ out, int n) {
    int i = blockIdx.x * 256 + threadIdx.x;
    if (i < n) out[i] = f2bf(in[i]);
}

// x [B,256,N] fp32 -> xcat channels 0..255 (batch stride 512*N), bf16
__global__ __launch_bounds__(256) void k_x_to_xcat(
    const float* __restrict__ x, unsigned short* __restrict__ xcat) {
    int i = blockIdx.x * 256 + threadIdx.x;              // total = B*256*N
    int b   = i / (DD * NN);
    int rem = i - b * (DD * NN);
    xcat[(size_t)b * D2 * NN + rem] = f2bf(x[i]);
}

// ---------------------------------------------------------------------------
// Generic WMMA GEMM:  Out[b][o][n] = sum_c W[o][c] * X[b][c][n] + bias[o]
// W bf16 [O,Cin] row-major, X bf16 (batch stride xBatch elements)
// Block tile: 64 O x 128 N, 8 waves, each wave 32x32 (2x2 WMMA tiles), K step 32
// A tile staged with global_load_async_to_lds_b128 (straight copy);
// B tile transpose-staged manually.
// ---------------------------------------------------------------------------
template <bool OUT_BF16>
__global__ __launch_bounds__(256) void k_gemm(
    const unsigned short* __restrict__ W, const unsigned short* __restrict__ X,
    const float* __restrict__ bias, void* __restrict__ Out,
    int O, int Cin, int Ncols, long long xBatch, long long oBatch) {
    __shared__ unsigned short lds_a[64][32];   // [o_local][k] k-contiguous
    __shared__ unsigned short lds_b[128][40];  // [n_local][k], pad->row 80B (16B-aligned)

    const int tid   = threadIdx.x;
    const int nbase = blockIdx.x * 128;
    const int obase = blockIdx.y * 64;
    const int b     = blockIdx.z;
    const size_t xoff = (size_t)b * (size_t)xBatch;
    const size_t ooff = (size_t)b * (size_t)oBatch;
    const int ln = tid & 15, hf = (tid >> 4) & 1, w = tid >> 5;
    const int wo = w >> 2, wn = w & 3;

    const v8f vz = {0.f,0.f,0.f,0.f,0.f,0.f,0.f,0.f};
    v8f acc[2][2] = {{vz, vz}, {vz, vz}};

    for (int k0 = 0; k0 < Cin; k0 += 32) {
        {   // stage A: 64 rows x 32 k, async 16B chunks straight into LDS
            int row = tid >> 2, ck = (tid & 3) << 3;
            async_load_b128(W + (size_t)(obase + row) * Cin + k0 + ck,
                            &lds_a[row][ck]);
        }
        {   // stage B transposed: global [c][n] -> lds [n][k]
            int kk = tid >> 3, nb = (tid & 7) << 4;
            union { uint4 q[2]; unsigned short s[16]; } u;
            const unsigned short* g = X + xoff + (size_t)(k0 + kk) * Ncols + nbase + nb;
            u.q[0] = ((const uint4*)g)[0];
            u.q[1] = ((const uint4*)g)[1];
#pragma unroll
            for (int j = 0; j < 16; ++j) lds_b[nb + j][kk] = u.s[j];
            if (k0 + 32 < Cin) {
                __builtin_prefetch(X + xoff + (size_t)(k0 + 32 + kk) * Ncols + nbase + nb, 0, 1);
            }
        }
        wait_async();
        __syncthreads();

        Frag16 af[2], bf[2];
#pragma unroll
        for (int ot = 0; ot < 2; ++ot) {
            const unsigned short* p = &lds_a[wo * 32 + ot * 16 + ln][8 * hf];
            af[ot].q[0] = *(const uint4*)p;
            af[ot].q[1] = *(const uint4*)(p + 16);
        }
#pragma unroll
        for (int nt = 0; nt < 2; ++nt) {
            const unsigned short* p = &lds_b[wn * 32 + nt * 16 + ln][8 * hf];
            bf[nt].q[0] = *(const uint4*)p;
            bf[nt].q[1] = *(const uint4*)(p + 16);
        }
#pragma unroll
        for (int ot = 0; ot < 2; ++ot)
#pragma unroll
            for (int nt = 0; nt < 2; ++nt)
                acc[ot][nt] = wmma_bf16(af[ot], bf[nt], acc[ot][nt]);
        __syncthreads();
    }

    // epilogue: D layout -> row = r + 8*(lane/16), col = lane%16
#pragma unroll
    for (int ot = 0; ot < 2; ++ot) {
#pragma unroll
        for (int nt = 0; nt < 2; ++nt) {
#pragma unroll
            for (int r = 0; r < 8; ++r) {
                int o = obase + wo * 32 + ot * 16 + r + 8 * hf;
                int n = nbase + wn * 32 + nt * 16 + ln;
                float val = acc[ot][nt][r] + bias[o];
                size_t idx = ooff + (size_t)o * Ncols + n;
                if constexpr (OUT_BF16)
                    ((unsigned short*)Out)[idx] = f2bf(val);
                else
                    ((float*)Out)[idx] = val;
            }
        }
    }
}

// ---------------------------------------------------------------------------
// Flash attention per (b,h): Q,K,V bf16 [B,256,len] (head rows h*64..h*64+63)
// One WG = 128 query rows; 8 waves x 16 rows; M tiled by 64; online softmax.
// V tile staged via async global->LDS (layout already matches B-fragments).
// msg output channel = dh*H + h (reference reshape order), coalesced 32B stores.
// ---------------------------------------------------------------------------
__global__ __launch_bounds__(256) void k_attention(
    const unsigned short* __restrict__ Q, const unsigned short* __restrict__ Kg,
    const unsigned short* __restrict__ V, unsigned short* __restrict__ msg) {
    __shared__ unsigned short lds_q[128][72];     // [n][dh] (transposed)
    __shared__ unsigned short lds_k[64][72];      // [m][dh] (transposed)
    __shared__ unsigned short lds_v[64][72];      // [dh][m] (natural)
    __shared__ unsigned short lds_p[8][16][72];   // per-wave strip [row][64]

    const int tid = threadIdx.x;
    const int b = blockIdx.z, h = blockIdx.y;
    const int n0 = blockIdx.x * 128;
    const int ln = tid & 15, hf = (tid >> 4) & 1, w = tid >> 5;
    const int lane = tid & 31;

    const unsigned short* Qb = Q  + ((size_t)b * DD + h * DHH) * NN;
    const unsigned short* Kb = Kg + ((size_t)b * DD + h * DHH) * MM;
    const unsigned short* Vb = V  + ((size_t)b * DD + h * DHH) * MM;

    {   // stage Q tile transposed: [dh][n] -> lds_q[n][dh]
        int dh = tid >> 2, nb = (tid & 3) * 32;
        union { uint4 q[4]; unsigned short s[32]; } u;
        const uint4* g = (const uint4*)(Qb + (size_t)dh * NN + n0 + nb);
        u.q[0] = g[0]; u.q[1] = g[1]; u.q[2] = g[2]; u.q[3] = g[3];
#pragma unroll
        for (int j = 0; j < 32; ++j) lds_q[nb + j][dh] = u.s[j];
    }
    __syncthreads();

    Frag16 qa[2];   // persistent Q A-fragments (dh k-chunks 0..31, 32..63)
#pragma unroll
    for (int ks = 0; ks < 2; ++ks) {
        const unsigned short* p = &lds_q[w * 16 + ln][ks * 32 + 8 * hf];
        qa[ks].q[0] = *(const uint4*)p;
        qa[ks].q[1] = *(const uint4*)(p + 16);
    }

    const v8f vz = {0.f,0.f,0.f,0.f,0.f,0.f,0.f,0.f};
    float rmax[8], rsum[8];
    v8f accm[4];
#pragma unroll
    for (int r = 0; r < 8; ++r) { rmax[r] = -1e30f; rsum[r] = 0.f; }
#pragma unroll
    for (int t = 0; t < 4; ++t) accm[t] = vz;

    const float scale = 0.125f;  // 1/sqrt(DH)

    for (int m0 = 0; m0 < MM; m0 += 64) {
        {   // stage V async (natural [dh][m]) and K manually (transposed)
            int dh = tid >> 2, mb = (tid & 3) * 16;
            async_load_b128(Vb + (size_t)dh * MM + m0 + mb, &lds_v[dh][mb]);
            async_load_b128(Vb + (size_t)dh * MM + m0 + mb + 8, &lds_v[dh][mb + 8]);
            union { uint4 q[2]; unsigned short s[16]; } u;
            const uint4* gk = (const uint4*)(Kb + (size_t)dh * MM + m0 + mb);
            u.q[0] = gk[0]; u.q[1] = gk[1];
#pragma unroll
            for (int j = 0; j < 16; ++j) lds_k[mb + j][dh] = u.s[j];
            if (m0 + 64 < MM) {
                __builtin_prefetch(Kb + (size_t)dh * MM + m0 + 64 + mb, 0, 1);
                __builtin_prefetch(Vb + (size_t)dh * MM + m0 + 64 + mb, 0, 1);
            }
        }
        wait_async();
        __syncthreads();

        // scores: 4 m-tiles x (2 wmma over dh)
        v8f s[4];
#pragma unroll
        for (int t = 0; t < 4; ++t) s[t] = vz;
#pragma unroll
        for (int mt = 0; mt < 4; ++mt) {
#pragma unroll
            for (int ks = 0; ks < 2; ++ks) {
                Frag16 kf;
                const unsigned short* p = &lds_k[mt * 16 + ln][ks * 32 + 8 * hf];
                kf.q[0] = *(const uint4*)p;
                kf.q[1] = *(const uint4*)(p + 16);
                s[mt] = wmma_bf16(qa[ks], kf, s[mt]);
            }
        }

        // online softmax update (row = r + 8*hf, col = lane%16 per m-tile)
#pragma unroll
        for (int r = 0; r < 8; ++r) {
            float mx = fmaxf(fmaxf(s[0][r], s[1][r]), fmaxf(s[2][r], s[3][r])) * scale;
#pragma unroll
            for (int off = 1; off < 16; off <<= 1) mx = fmaxf(mx, __shfl_xor(mx, off));
            float mn  = fmaxf(rmax[r], mx);
            float fac = __expf(rmax[r] - mn);
            rmax[r] = mn;
            float psum = 0.f;
#pragma unroll
            for (int t = 0; t < 4; ++t) {
                float p = __expf(s[t][r] * scale - mn);
                psum += p;
                accm[t][r] *= fac;
                lds_p[w][r + 8 * hf][t * 16 + ln] = f2bf(p);
            }
            rsum[r] = rsum[r] * fac + psum;
        }

        // msg += P x V^T  (wave-private P strip; no cross-wave barrier needed)
#pragma unroll
        for (int ks = 0; ks < 2; ++ks) {
            Frag16 pf;
            const unsigned short* p = &lds_p[w][ln][ks * 32 + 8 * hf];
            pf.q[0] = *(const uint4*)p;
            pf.q[1] = *(const uint4*)(p + 16);
#pragma unroll
            for (int dt = 0; dt < 4; ++dt) {
                Frag16 vf;
                const unsigned short* pv = &lds_v[dt * 16 + ln][ks * 32 + 8 * hf];
                vf.q[0] = *(const uint4*)pv;
                vf.q[1] = *(const uint4*)(pv + 16);
                accm[dt] = wmma_bf16(pf, vf, accm[dt]);
            }
        }
        __syncthreads();
    }

    // finalize: divide by row sums; transpose through wave-private strip so
    // each lane emits one contiguous 32B store per channel row.
#pragma unroll
    for (int r = 0; r < 8; ++r) {
        float tot = rsum[r];
#pragma unroll
        for (int off = 1; off < 16; off <<= 1) tot += __shfl_xor(tot, off);
        float inv = 1.f / tot;
#pragma unroll
        for (int dt = 0; dt < 4; ++dt)
            lds_p[w][r + 8 * hf][dt * 16 + ln] = f2bf(accm[dt][r] * inv);
    }
#pragma unroll
    for (int rep = 0; rep < 2; ++rep) {
        int dh = rep * 32 + lane;
        union { uint4 q[2]; unsigned short s[16]; } u;
#pragma unroll
        for (int j = 0; j < 16; ++j) u.s[j] = lds_p[w][j][dh];
        uint4* gp = (uint4*)(msg + ((size_t)b * DD + dh * HH + h) * NN + n0 + w * 16);
        gp[0] = u.q[0];
        gp[1] = u.q[1];
    }
}

// ---------------------------------------------------------------------------
// InstanceNorm over N per (b,c) + ReLU, fp32 in -> bf16 out. One WG per (b,c).
// ---------------------------------------------------------------------------
__global__ __launch_bounds__(256) void k_instnorm(
    const float* __restrict__ h1, unsigned short* __restrict__ h2) {
    __shared__ float s1[256], s2[256];
    const size_t base = (size_t)blockIdx.x * NN;
    float a = 0.f, bb = 0.f;
    for (int i = threadIdx.x; i < NN; i += 256) {
        float v = h1[base + i];
        a += v; bb += v * v;
    }
    s1[threadIdx.x] = a; s2[threadIdx.x] = bb;
    __syncthreads();
    for (int st = 128; st > 0; st >>= 1) {
        if (threadIdx.x < st) {
            s1[threadIdx.x] += s1[threadIdx.x + st];
            s2[threadIdx.x] += s2[threadIdx.x + st];
        }
        __syncthreads();
    }
    float mean = s1[0] * (1.f / NN);
    float var  = s2[0] * (1.f / NN) - mean * mean;
    float rstd = rsqrtf(var + 1e-5f);
    for (int i = threadIdx.x; i < NN; i += 256) {
        float v = (h1[base + i] - mean) * rstd;
        h2[base + i] = f2bf(v > 0.f ? v : 0.f);
    }
}

// ---------------------------------------------------------------------------
extern "C" void kernel_launch(void* const* d_in, const int* in_sizes, int n_in,
                              void* d_out, int out_size, void* d_ws, size_t ws_size,
                              hipStream_t stream) {
    (void)in_sizes; (void)n_in; (void)out_size; (void)ws_size;
    const float* x   = (const float*)d_in[0];
    const float* src = (const float*)d_in[1];
    const float* Wq  = (const float*)d_in[2];
    const float* bq  = (const float*)d_in[3];
    const float* Wk  = (const float*)d_in[4];
    const float* bk  = (const float*)d_in[5];
    const float* Wv  = (const float*)d_in[6];
    const float* bv  = (const float*)d_in[7];
    const float* Wm  = (const float*)d_in[8];
    const float* bm  = (const float*)d_in[9];
    const float* W1  = (const float*)d_in[10];
    const float* b1  = (const float*)d_in[11];
    const float* W2  = (const float*)d_in[12];
    const float* b2  = (const float*)d_in[13];
    float* out = (float*)d_out;

    char* ws = (char*)d_ws;
    size_t off = 0;
    auto carve = [&](size_t bytes) -> char* {
        char* p = ws + off;
        off += (bytes + 255) & ~(size_t)255;
        return p;
    };

    unsigned short* WqB  = (unsigned short*)carve((size_t)DD * DD * 2);
    unsigned short* WkB  = (unsigned short*)carve((size_t)DD * DD * 2);
    unsigned short* WvB  = (unsigned short*)carve((size_t)DD * DD * 2);
    unsigned short* WmB  = (unsigned short*)carve((size_t)DD * DD * 2);
    unsigned short* W1B  = (unsigned short*)carve((size_t)D2 * D2 * 2);
    unsigned short* W2B  = (unsigned short*)carve((size_t)DD * D2 * 2);
    unsigned short* xcat = (unsigned short*)carve((size_t)BB * D2 * NN * 2); // [x ; message]
    unsigned short* srcB = (unsigned short*)carve((size_t)BB * DD * MM * 2);
    unsigned short* Qb   = (unsigned short*)carve((size_t)BB * DD * NN * 2);
    unsigned short* Kb   = (unsigned short*)carve((size_t)BB * DD * MM * 2);
    unsigned short* Vb   = (unsigned short*)carve((size_t)BB * DD * MM * 2);
    unsigned short* msgB = (unsigned short*)carve((size_t)BB * DD * NN * 2);
    float*          h1   = (float*)         carve((size_t)BB * D2 * NN * 4);
    unsigned short* h2   = (unsigned short*)carve((size_t)BB * D2 * NN * 2);

    dim3 blk(256);
    auto cgrid = [](int n) { return dim3((unsigned)((n + 255) / 256)); };

    // fp32 -> bf16 conversions
    k_f32_to_bf16<<<cgrid(DD * DD), blk, 0, stream>>>(Wq, WqB, DD * DD);
    k_f32_to_bf16<<<cgrid(DD * DD), blk, 0, stream>>>(Wk, WkB, DD * DD);
    k_f32_to_bf16<<<cgrid(DD * DD), blk, 0, stream>>>(Wv, WvB, DD * DD);
    k_f32_to_bf16<<<cgrid(DD * DD), blk, 0, stream>>>(Wm, WmB, DD * DD);
    k_f32_to_bf16<<<cgrid(D2 * D2), blk, 0, stream>>>(W1, W1B, D2 * D2);
    k_f32_to_bf16<<<cgrid(DD * D2), blk, 0, stream>>>(W2, W2B, DD * D2);
    k_f32_to_bf16<<<cgrid(BB * DD * MM), blk, 0, stream>>>(src, srcB, BB * DD * MM);
    k_x_to_xcat<<<cgrid(BB * DD * NN), blk, 0, stream>>>(x, xcat);

    dim3 gq(NN / 128, DD / 64, BB);   // (16, 4, 8)
    dim3 g1(NN / 128, D2 / 64, BB);   // (16, 8, 8)

    // Q = Wq @ x (+bq), K = Wk @ src, V = Wv @ src  -> bf16
    k_gemm<true><<<gq, blk, 0, stream>>>(WqB, xcat, bq, Qb, DD, DD, NN,
                                         (long long)D2 * NN, (long long)DD * NN);
    k_gemm<true><<<gq, blk, 0, stream>>>(WkB, srcB, bk, Kb, DD, DD, MM,
                                         (long long)DD * MM, (long long)DD * MM);
    k_gemm<true><<<gq, blk, 0, stream>>>(WvB, srcB, bv, Vb, DD, DD, MM,
                                         (long long)DD * MM, (long long)DD * MM);

    // flash attention -> msg (channel order dh*H + h)
    k_attention<<<dim3(NN / 128, HH, BB), blk, 0, stream>>>(Qb, Kb, Vb, msgB);

    // message = Wm @ msg (+bm) -> upper 256 channels of xcat
    k_gemm<true><<<gq, blk, 0, stream>>>(WmB, msgB, bm, xcat + (size_t)DD * NN,
                                         DD, DD, NN, (long long)DD * NN,
                                         (long long)D2 * NN);

    // h1 = W1 @ [x ; message] (+b1) -> fp32 (for exact IN stats)
    k_gemm<false><<<g1, blk, 0, stream>>>(W1B, xcat, b1, h1, D2, D2, NN,
                                          (long long)D2 * NN, (long long)D2 * NN);

    // InstanceNorm + ReLU -> bf16
    k_instnorm<<<dim3(BB * D2), blk, 0, stream>>>(h1, h2);

    // out = W2 @ h2 (+b2) -> fp32 d_out
    k_gemm<false><<<gq, blk, 0, stream>>>(W2B, h2, b2, out, DD, D2, NN,
                                          (long long)D2 * NN, (long long)DD * NN);
}
